// GAT_2499670966385
// MI455X (gfx1250) — matrix operand order repaired and checked by reading, not compiled, not run on hardware
//
#include <hip/hip_runtime.h>
#include <hip/hip_bf16.h>

// ---------------------------------------------------------------------------
// GAT-style model on MI455X (gfx1250, wave32, WMMA 16x16x32 f16).
// N=8192, T=60, D=6, H=64, G=3H=192.
// ---------------------------------------------------------------------------

#define NN 8192
#define TT 60
#define DDIM 6
#define HH 64
#define GG 192
#define NEG_SLOPE 0.01f

typedef __attribute__((ext_vector_type(16))) _Float16 v16h;
typedef __attribute__((ext_vector_type(8)))  float    v8f;

__device__ __forceinline__ v8f z8() {
    v8f z = {0.f,0.f,0.f,0.f,0.f,0.f,0.f,0.f};
    return z;
}

__device__ __forceinline__ v8f wmma_f16(v16h a, v16h b, v8f c) {
    // D = A(16x32 f16) * B(32x16 f16) + C(16x16 f32)
    return __builtin_amdgcn_wmma_f32_16x16x32_f16(
        false, a, false, b, (short)0, c, false, false);
}

// ---- A-fragment (16x32 f16) element mapping (ISA 7.12.2) -------------------
// lanes 0-15 cover K {0..7,16..23}, lanes 16-31 cover K {8..15,24..31};
// element e<8 -> low K group, e>=8 -> +16.
__device__ __forceinline__ int a_hi(int kk) { return (kk >> 3) & 1; }
__device__ __forceinline__ int a_el(int kk) { return (kk & 7) | ((kk >> 4) << 3); }
__device__ __forceinline__ int a_kk(int lane, int e) {
    return (e & 7) + ((e >> 3) << 4) + ((lane >> 4) << 3);
}
// ---- B-fragment (32x16 f16): same K mapping as A (lanes 0-15 K {0..7,16..23}, lanes 16-31 K {8..15,24..31}).
__device__ __forceinline__ int b_hi(int kk) { return (kk >> 3) & 1; }
__device__ __forceinline__ int b_el(int kk) { return (kk & 7) | ((kk >> 4) << 3); }

__device__ __forceinline__ float sigm(float x) { return 1.f / (1.f + __expf(-x)); }
__device__ __forceinline__ float fast_tanh(float x) {
    float e = __expf(2.f * x);
    return (e - 1.f) / (e + 1.f);
}
__device__ __forceinline__ float lrelu(float x) { return x > 0.f ? x : NEG_SLOPE * x; }

// B-fragment linear index: [(kt*12 + nt)*32 + lane]*16 + e
__device__ __forceinline__ v16h ldfrag(const _Float16* base, int kt, int nt, int lane) {
    return *(const v16h*)(base + ((((kt * 12 + nt) * 32) + lane) << 4));
}

// ===========================================================================
// Kernel 0: pack Whh0 / Wih1 / Whh1 (each 192x64 f32, row-major, gemm is
// h @ W.T so B[k][g] = W[g][k]) into f16 B-fragment layout in workspace.
// ===========================================================================
__global__ void k_init_wfrags(const float* __restrict__ Whh0,
                              const float* __restrict__ Wih1,
                              const float* __restrict__ Whh1,
                              _Float16* __restrict__ dst) {
    int idx = blockIdx.x * 256 + threadIdx.x;           // 3 * 192*64
    if (idx >= 3 * GG * HH) return;
    int m = idx / (GG * HH), r = idx % (GG * HH);
    const float* W = (m == 0) ? Whh0 : ((m == 1) ? Wih1 : Whh1);
    _Float16* d = dst + m * (GG * HH);
    int g = r / HH, k = r % HH;
    int kt = k >> 5, kk = k & 31;
    int lane = (g & 15) + (b_hi(kk) << 4);
    int nt = g >> 4;
    d[(((kt * 12 + nt) * 32 + lane) << 4) + b_el(kk)] = (_Float16)W[g * HH + k];
}

// ===========================================================================
// Kernel 1: fused 2-layer GRU. 64 blocks x 256 threads (8 waves).
// Each wave owns 16 batch rows; h kept as f16 A-fragments in LDS.
// All weight B-fragments staged in LDS (72 KB; gfx1250 has 320 KB/WGP):
// steady-state recurrence touches no global memory for weights.
// ===========================================================================
__global__ void __launch_bounds__(256) k_gru(
    const float* __restrict__ x,
    const float* __restrict__ Wih0,
    const float* __restrict__ bih0, const float* __restrict__ bhh0,
    const float* __restrict__ bih1, const float* __restrict__ bhh1,
    const _Float16* __restrict__ wfrag,
    float* __restrict__ hidden) {

    __shared__ __align__(32) _Float16 sWF[3 * GG * HH];   // 72 KB packed B-frags
    __shared__ __align__(32) _Float16 aF0[8][2][32][16];  // layer-0 h as A-frags per wave
    __shared__ __align__(32) _Float16 aF1[8][2][32][16];  // layer-1 h
    __shared__ float ldsX[128][DDIM];
    __shared__ float sWih0[GG * DDIM];
    __shared__ float sB[4][GG];                           // bih0,bhh0,bih1,bhh1

    const int tid = threadIdx.x;
    const int w = tid >> 5, lane = tid & 31;
    const int Rb = blockIdx.x * 128;
    const int Rw = Rb + w * 16;
    const int n16 = lane & 15, hi16 = lane >> 4;

    // stage weights / biases into LDS
    {
        const uint32_t* src = (const uint32_t*)wfrag;
        uint32_t* dstw = (uint32_t*)sWF;
        for (int i = tid; i < (3 * GG * HH) / 2; i += 256) dstw[i] = src[i];
    }
    for (int i = tid; i < GG * DDIM; i += 256) sWih0[i] = Wih0[i];
    for (int i = tid; i < GG; i += 256) {
        sB[0][i] = bih0[i]; sB[1][i] = bhh0[i];
        sB[2][i] = bih1[i]; sB[3][i] = bhh1[i];
    }
    for (int i = tid; i < 8 * 2 * 32 * 16; i += 256) {
        ((_Float16*)aF0)[i] = (_Float16)0.f;
        ((_Float16*)aF1)[i] = (_Float16)0.f;
    }
    __syncthreads();

    const _Float16* fWhh0 = sWF;
    const _Float16* fWih1 = sWF + GG * HH;
    const _Float16* fWhh1 = sWF + 2 * GG * HH;

    for (int t = 0; t < TT; ++t) {
        // stage x[:, d*T + t] for this block's 128 rows
        for (int i = tid; i < 128 * DDIM; i += 256) {
            int row = i / DDIM, d = i % DDIM;
            ldsX[row][d] = x[(Rb + row) * (DDIM * TT) + d * TT + t];
        }
        __syncthreads();

        // ---------------- layer 0 ----------------
        v16h a0k0 = *(const v16h*)&aF0[w][0][lane][0];
        v16h a0k1 = *(const v16h*)&aF0[w][1][lane][0];
        float hnew0[4][8];

        for (int ct = 0; ct < 4; ++ct) {
            v8f gr = z8(), gz = z8(), gn = z8();
            gr = wmma_f16(a0k0, ldfrag(fWhh0, 0, ct,     lane), gr);
            gr = wmma_f16(a0k1, ldfrag(fWhh0, 1, ct,     lane), gr);
            gz = wmma_f16(a0k0, ldfrag(fWhh0, 0, ct + 4, lane), gz);
            gz = wmma_f16(a0k1, ldfrag(fWhh0, 1, ct + 4, lane), gz);
            gn = wmma_f16(a0k0, ldfrag(fWhh0, 0, ct + 8, lane), gn);
            gn = wmma_f16(a0k1, ldfrag(fWhh0, 1, ct + 8, lane), gn);

            // input path (K=6) via VALU
            int g = ct * 16 + n16;
            float wR[DDIM], wZ[DDIM], wNv[DDIM];
#pragma unroll
            for (int d = 0; d < DDIM; ++d) {
                wR[d]  = sWih0[g * DDIM + d];
                wZ[d]  = sWih0[(g + 64) * DDIM + d];
                wNv[d] = sWih0[(g + 128) * DDIM + d];
            }
            float biR = sB[0][g], biZ = sB[0][g + 64], biN = sB[0][g + 128];
            float bhR = sB[1][g], bhZ = sB[1][g + 64], bhN = sB[1][g + 128];
#pragma unroll
            for (int v = 0; v < 8; ++v) {
                int M = v + 8 * hi16;
                float giR = biR, giZ = biZ, giN = biN;
#pragma unroll
                for (int d = 0; d < DDIM; ++d) {
                    float xv = ldsX[w * 16 + M][d];
                    giR += xv * wR[d]; giZ += xv * wZ[d]; giN += xv * wNv[d];
                }
                int col = ct * 16 + n16, kk = col & 31, kt = col >> 5;
                float hold = (float)aF0[w][kt][M + 16 * a_hi(kk)][a_el(kk)];
                float r  = sigm(giR + gr[v] + bhR);
                float zz = sigm(giZ + gz[v] + bhZ);
                float nn = fast_tanh(giN + r * (gn[v] + bhN));
                hnew0[ct][v] = (1.f - zz) * nn + zz * hold;
            }
        }
        // scatter h0_new back into A-fragment layout (f16)
#pragma unroll
        for (int ct = 0; ct < 4; ++ct)
#pragma unroll
            for (int v = 0; v < 8; ++v) {
                int M = v + 8 * hi16, col = ct * 16 + n16, kk = col & 31, kt = col >> 5;
                aF0[w][kt][M + 16 * a_hi(kk)][a_el(kk)] = (_Float16)hnew0[ct][v];
            }

        // ---------------- layer 1 ----------------
        v16h ain0 = *(const v16h*)&aF0[w][0][lane][0];   // input = h0_new
        v16h ain1 = *(const v16h*)&aF0[w][1][lane][0];
        v16h a1k0 = *(const v16h*)&aF1[w][0][lane][0];
        v16h a1k1 = *(const v16h*)&aF1[w][1][lane][0];
        float hnew1[4][8];

        for (int ct = 0; ct < 4; ++ct) {
            v8f accR = z8(), accZ = z8(), giN = z8(), ghN = z8();
            accR = wmma_f16(ain0, ldfrag(fWih1, 0, ct,     lane), accR);
            accR = wmma_f16(ain1, ldfrag(fWih1, 1, ct,     lane), accR);
            accR = wmma_f16(a1k0, ldfrag(fWhh1, 0, ct,     lane), accR);
            accR = wmma_f16(a1k1, ldfrag(fWhh1, 1, ct,     lane), accR);
            accZ = wmma_f16(ain0, ldfrag(fWih1, 0, ct + 4, lane), accZ);
            accZ = wmma_f16(ain1, ldfrag(fWih1, 1, ct + 4, lane), accZ);
            accZ = wmma_f16(a1k0, ldfrag(fWhh1, 0, ct + 4, lane), accZ);
            accZ = wmma_f16(a1k1, ldfrag(fWhh1, 1, ct + 4, lane), accZ);
            giN  = wmma_f16(ain0, ldfrag(fWih1, 0, ct + 8, lane), giN);
            giN  = wmma_f16(ain1, ldfrag(fWih1, 1, ct + 8, lane), giN);
            ghN  = wmma_f16(a1k0, ldfrag(fWhh1, 0, ct + 8, lane), ghN);
            ghN  = wmma_f16(a1k1, ldfrag(fWhh1, 1, ct + 8, lane), ghN);

            int g = ct * 16 + n16;
            float bR = sB[2][g] + sB[3][g];
            float bZ = sB[2][g + 64] + sB[3][g + 64];
            float biN = sB[2][g + 128], bhN = sB[3][g + 128];
#pragma unroll
            for (int v = 0; v < 8; ++v) {
                int M = v + 8 * hi16;
                int col = ct * 16 + n16, kk = col & 31, kt = col >> 5;
                float hold = (float)aF1[w][kt][M + 16 * a_hi(kk)][a_el(kk)];
                float r  = sigm(accR[v] + bR);
                float zz = sigm(accZ[v] + bZ);
                float nn = fast_tanh(giN[v] + biN + r * (ghN[v] + bhN));
                hnew1[ct][v] = (1.f - zz) * nn + zz * hold;
            }
        }
#pragma unroll
        for (int ct = 0; ct < 4; ++ct)
#pragma unroll
            for (int v = 0; v < 8; ++v) {
                int M = v + 8 * hi16, col = ct * 16 + n16, kk = col & 31, kt = col >> 5;
                aF1[w][kt][M + 16 * a_hi(kk)][a_el(kk)] = (_Float16)hnew1[ct][v];
            }

        __syncthreads();   // protect ldsX restage
    }

    // write final hidden state (f32)
#pragma unroll
    for (int ct = 0; ct < 4; ++ct)
#pragma unroll
        for (int v = 0; v < 8; ++v) {
            int M = v + 8 * hi16, col = ct * 16 + n16, kk = col & 31, kt = col >> 5;
            hidden[(Rw + M) * HH + col] = (float)aF1[w][kt][M + 16 * a_hi(kk)][a_el(kk)];
        }
}

// ===========================================================================
// Kernel 2: t = hidden @ transfer_W.T + b ;  s1 = t@a1, s2 = t@a2.
// One thread per row. 32 blocks x 256.
// ===========================================================================
__global__ void __launch_bounds__(256) k_transfer(
    const float* __restrict__ hidden, const float* __restrict__ tW,
    const float* __restrict__ tb, const float* __restrict__ a,
    float* __restrict__ s1, float* __restrict__ s2) {
    int r = blockIdx.x * 256 + threadIdx.x;
    float h[HH];
#pragma unroll
    for (int k = 0; k < HH; ++k) h[k] = hidden[r * HH + k];
    float acc1 = 0.f, acc2 = 0.f;
    for (int c = 0; c < HH; ++c) {
        float tv = tb[c];
#pragma unroll
        for (int k = 0; k < HH; ++k) tv += tW[c * HH + k] * h[k];
        acc1 += tv * a[c];
        acc2 += tv * a[HH + c];
    }
    s1[r] = acc1; s2[r] = acc2;
}

// ===========================================================================
// Kernel 3: global max of s1 (leaky-relu is monotonic -> softmax row max
// is lr(s2[i] + max_j s1[j])). Single block.
// ===========================================================================
__global__ void k_s1max(const float* __restrict__ s1, float* __restrict__ s1max) {
    __shared__ float red[256];
    int tid = threadIdx.x;
    float m = -1e30f;
    for (int i = tid; i < NN; i += 256) m = fmaxf(m, s1[i]);
    red[tid] = m; __syncthreads();
    for (int s = 128; s > 0; s >>= 1) {
        if (tid < s) red[tid] = fmaxf(red[tid], red[tid + s]);
        __syncthreads();
    }
    if (tid == 0) s1max[0] = red[0];
}

// ===========================================================================
// Kernel 4: pack hidden (8192x64 f32) into f16 B-fragment layout:
// [jchunk(256)][ct(4)][lane(32)][e(16)].
// ===========================================================================
__global__ void k_hb(const float* __restrict__ hidden, _Float16* __restrict__ hb) {
    int idx = blockIdx.x * 256 + threadIdx.x;      // 8192*64
    int j = idx >> 6, col = idx & 63;
    int jc = j >> 5, kk = j & 31, ct = col >> 4, n = col & 15;
    hb[(((jc * 4 + ct) * 32 + n + (b_hi(kk) << 4)) << 4) + b_el(kk)] = (_Float16)hidden[idx];
}

// ===========================================================================
// Kernel 5: streaming attention (never materializes 8192^2) + fc + out.
// 64 blocks x 256 threads; each wave owns 16 query rows, loops all 8192 keys
// in K=32 chunks; weight tile -> f16 A-fragment -> WMMA against hiddenB.
// global_prefetch keeps the hiddenB stream ahead of consumption.
// ===========================================================================
__global__ void __launch_bounds__(256) k_attn(
    const float* __restrict__ hidden, const _Float16* __restrict__ hiddenB,
    const float* __restrict__ s1g, const float* __restrict__ s2g,
    const float* __restrict__ s1maxp,
    const float* __restrict__ fcW, const float* __restrict__ fcb,
    const float* __restrict__ outW, const float* __restrict__ outbp,
    float* __restrict__ out) {

    __shared__ float sm[NN + HH * HH + HH + HH + 128];  // s1 (reused as hAtt) | fcW | fcb | outW | den
    float* sS1  = sm;
    float* sFcW = sm + NN;
    float* sFcB = sFcW + HH * HH;
    float* sOutW = sFcB + HH;
    float* sDen = sOutW + HH;

    const int tid = threadIdx.x, w = tid >> 5, lane = tid & 31;
    const int Rw = blockIdx.x * 128 + w * 16;
    const int n16 = lane & 15, hi16 = lane >> 4;

    for (int i = tid; i < NN; i += 256) sS1[i] = s1g[i];
    for (int i = tid; i < HH * HH; i += 256) sFcW[i] = fcW[i];
    if (tid < HH) { sFcB[tid] = fcb[tid]; sOutW[tid] = outW[tid]; }
    __syncthreads();

    const float s1max = s1maxp[0];
    const float s2i = s2g[Rw + n16];                 // this lane's query row
    const float mi = lrelu(s2i + s1max);             // exact row max of scores

    v8f C0 = z8(), C1 = z8(), C2 = z8(), C3 = z8();
    float den = 0.f;

    for (int jc = 0; jc < NN / 32; ++jc) {
        // prefetch the chunk 4 iterations ahead (gfx1250 global_prefetch path)
        if (jc + 4 < NN / 32)
            __builtin_prefetch((const void*)(hiddenB + (jc + 4) * 2048 + lane * 16), 0, 0);

        v16h af;
#pragma unroll
        for (int e = 0; e < 16; ++e) {
            int kk = a_kk(lane, e);
            float sc = s2i + sS1[jc * 32 + kk];
            float wv = __expf(lrelu(sc) - mi);
            den += wv;
            af[e] = (_Float16)wv;
        }
        const _Float16* bp = hiddenB + jc * 2048 + lane * 16;
        C0 = wmma_f16(af, *(const v16h*)(bp),        C0);
        C1 = wmma_f16(af, *(const v16h*)(bp + 512),  C1);
        C2 = wmma_f16(af, *(const v16h*)(bp + 1024), C2);
        C3 = wmma_f16(af, *(const v16h*)(bp + 1536), C3);
    }
    // lanes L and L+16 together cover the full K range of row M=L%16
    den += __shfl_xor(den, 16);
    if (hi16 == 0) sDen[w * 16 + n16] = den;
    __syncthreads();                                  // also fences sS1 reuse

    float* sH = sS1;                                  // reuse as hAtt [128][64]
    v8f Cs0 = C0, Cs1 = C1, Cs2 = C2, Cs3 = C3;
#pragma unroll
    for (int ct = 0; ct < 4; ++ct) {
        v8f Cc = (ct == 0) ? Cs0 : (ct == 1) ? Cs1 : (ct == 2) ? Cs2 : Cs3;
#pragma unroll
        for (int v = 0; v < 8; ++v) {
            int M = v + 8 * hi16;
            float dd = sDen[w * 16 + M];
            float hv = hidden[(Rw + M) * HH + ct * 16 + n16];
            sH[(w * 16 + M) * HH + ct * 16 + n16] = Cc[v] / dd + hv;
        }
    }
    __syncthreads();

    // fc (leaky) + out: 2 threads per row, 32 output cols each
    int row = tid >> 1, halfc = tid & 1;
    float hr[HH];
#pragma unroll
    for (int k = 0; k < HH; ++k) hr[k] = sH[row * HH + k];
    float s = 0.f;
    for (int c = halfc * 32; c < halfc * 32 + 32; ++c) {
        float acc = sFcB[c];
#pragma unroll
        for (int k = 0; k < HH; ++k) acc += sFcW[c * HH + k] * hr[k];
        s += lrelu(acc) * sOutW[c];
    }
    s += __shfl_xor(s, 1);
    if (halfc == 0) out[blockIdx.x * 128 + row] = s + outbp[0];
}

// ===========================================================================
extern "C" void kernel_launch(void* const* d_in, const int* in_sizes, int n_in,
                              void* d_out, int out_size, void* d_ws, size_t ws_size,
                              hipStream_t stream) {
    const float* x     = (const float*)d_in[0];
    const float* Wih0  = (const float*)d_in[1];
    const float* Whh0  = (const float*)d_in[2];
    const float* bih0  = (const float*)d_in[3];
    const float* bhh0  = (const float*)d_in[4];
    const float* Wih1  = (const float*)d_in[5];
    const float* Whh1  = (const float*)d_in[6];
    const float* bih1  = (const float*)d_in[7];
    const float* bhh1  = (const float*)d_in[8];
    const float* tW    = (const float*)d_in[9];
    const float* tb    = (const float*)d_in[10];
    const float* a     = (const float*)d_in[11];
    const float* fcW   = (const float*)d_in[12];
    const float* fcb   = (const float*)d_in[13];
    const float* outW  = (const float*)d_in[14];
    const float* outb  = (const float*)d_in[15];

    char* ws = (char*)d_ws;
    float*     hidden = (float*)(ws);                                   // 2 MB
    float*     s1     = (float*)(ws + (2u << 20));                      // 32 KB
    float*     s2     = (float*)(ws + (2u << 20) + (32u << 10));        // 32 KB
    float*     s1m    = (float*)(ws + (2u << 20) + (64u << 10));        // 4 B
    _Float16*  wfrag  = (_Float16*)(ws + (2u << 20) + (128u << 10));    // 72 KB
    _Float16*  hb     = (_Float16*)(ws + (2u << 20) + (256u << 10));    // 1 MB

    k_init_wfrags<<<144, 256, 0, stream>>>(Whh0, Wih1, Whh1, wfrag);
    k_gru<<<NN / 128, 256, 0, stream>>>(x, Wih0, bih0, bhh0, bih1, bhh1, wfrag, hidden);
    k_transfer<<<NN / 256, 256, 0, stream>>>(hidden, tW, tb, a, s1, s2);
    k_s1max<<<1, 256, 0, stream>>>(s1, s1m);
    k_hb<<<(NN * HH) / 256, 256, 0, stream>>>(hidden, hb);
    k_attn<<<NN / 128, 256, 0, stream>>>(hidden, hb, s1, s2, s1m,
                                         fcW, fcb, outW, outb, (float*)d_out);
}
